// EnhancedMeshEncoder_5385888989261
// MI455X (gfx1250) — matrix operand order repaired and checked by reading, not compiled, not run on hardware
//
#include <hip/hip_runtime.h>
#include <hip/hip_bf16.h>

#define NN 50000
#define EE 800000
#define TT (EE + NN)   // edges + self loops

typedef __attribute__((ext_vector_type(16))) _Float16 v16h;
typedef __attribute__((ext_vector_type(8)))  _Float16 v8h;
typedef __attribute__((ext_vector_type(8)))  float    v8f;

// ---------- helpers ----------
__device__ __forceinline__ unsigned fkey(float f) {
    unsigned u = __float_as_uint(f);
    return (u & 0x80000000u) ? ~u : (u | 0x80000000u);   // order-preserving map
}
__device__ __forceinline__ float finv(unsigned k) {
    return __uint_as_float((k & 0x80000000u) ? (k & 0x7fffffffu) : ~k);
}
__device__ __forceinline__ float leaky(float x) { return x > 0.f ? x : 0.2f * x; }

// ---------- degree / normalization ----------
__global__ void deg_count(const int* __restrict__ ei, float* __restrict__ deg) {
    int e = blockIdx.x * blockDim.x + threadIdx.x;
    if (e >= TT) return;
    int dst = (e < EE) ? ei[EE + e] : (e - EE);
    atomicAdd(&deg[dst], 1.0f);
}
__global__ void dinv_kernel(float* __restrict__ deg) {
    int i = blockIdx.x * blockDim.x + threadIdx.x;
    if (i < NN) deg[i] = rsqrtf(fmaxf(deg[i], 1.0f));
}

// ---------- GCN1: x[N,3] @ W1[3,64] ----------
__global__ void gcn1_mm(const float* __restrict__ x, const float* __restrict__ W1,
                        float* __restrict__ t1) {
    int i = blockIdx.x * blockDim.x + threadIdx.x;
    if (i >= NN * 64) return;
    int n = i >> 6, c = i & 63;
    const float* xp = x + (size_t)n * 3;
    t1[i] = xp[0] * W1[c] + xp[1] * W1[64 + c] + xp[2] * W1[128 + c];
}

// ---------- symmetric-normalized scatter (GCN message passing) ----------
__global__ void gcn_scatter(const float* __restrict__ t, const int* __restrict__ ei,
                            const float* __restrict__ dinv, float* __restrict__ out, int C) {
    long long i = (long long)blockIdx.x * blockDim.x + threadIdx.x;
    if (i >= (long long)TT * C) return;
    int e = (int)(i / C), c = (int)(i % C);
    int src = (e < EE) ? ei[e] : (e - EE);
    int dst = (e < EE) ? ei[EE + e] : (e - EE);
    float nrm = dinv[src] * dinv[dst];
    atomicAdd(&out[(size_t)dst * C + c], nrm * t[(size_t)src * C + c]);
}

// ---------- bias (+optional ReLU) + LayerNorm, emit f32 (optional) + f16 ----------
__global__ void postln(const float* __restrict__ acc, const float* __restrict__ bias,
                       const float* __restrict__ gamma, const float* __restrict__ beta,
                       float* __restrict__ out32, _Float16* __restrict__ out16,
                       int C, int do_relu) {
    __shared__ float s[256];
    int n = blockIdx.x, c = threadIdx.x;
    float v = acc[(size_t)n * C + c] + bias[c];
    if (do_relu) v = fmaxf(v, 0.f);
    s[c] = v; __syncthreads();
    for (int st = C >> 1; st > 0; st >>= 1) { if (c < st) s[c] += s[c + st]; __syncthreads(); }
    float mu = s[0] / (float)C; __syncthreads();
    float d = v - mu;
    s[c] = d * d; __syncthreads();
    for (int st = C >> 1; st > 0; st >>= 1) { if (c < st) s[c] += s[c + st]; __syncthreads(); }
    float r = rsqrtf(s[0] / (float)C + 1e-5f);
    float o = gamma[c] * d * r + beta[c];
    if (out32) out32[(size_t)n * C + c] = o;
    out16[(size_t)n * C + c] = (_Float16)o;
}

// ---------- f32 weight [K,Nc] -> f16 transposed [Nc,K] ----------
__global__ void cvt_tr_f16(const float* __restrict__ in, _Float16* __restrict__ out,
                           int K, int Nc) {
    int i = blockIdx.x * blockDim.x + threadIdx.x;
    if (i >= K * Nc) return;
    int k = i / Nc, c = i % Nc;
    out[(size_t)c * K + k] = (_Float16)in[i];
}

// ---------- WMMA GEMM: C[M,Nc] = A[M,K](f16,rm) x Bt[Nc,K](f16, B transposed) ----------
// one wave computes a 16x64 strip (4 n-tiles); K mult of 32, Nc mult of 64, M mult of 16
__global__ __launch_bounds__(32)
void wmma_gemm4(const _Float16* __restrict__ A, const _Float16* __restrict__ Bt,
                float* __restrict__ C, int K, int Nc) {
    int m0 = blockIdx.x * 16;
    int n0 = blockIdx.y * 64;
    int lane = threadIdx.x;
    int l15  = lane & 15;
    int hi   = lane >> 4;            // 0: lanes 0-15, 1: lanes 16-31
    v8f acc0 = {}, acc1 = {}, acc2 = {}, acc3 = {};
    const _Float16* arow = A + (size_t)(m0 + l15) * K + hi * 16;
    const _Float16* brow = Bt + (size_t)(n0 + l15) * K + hi * 16;
    for (int k0 = 0; k0 < K; k0 += 32) {
        // A frag: lanes 0-15 K k0..k0+15; lanes 16-31 +16
        v16h af = *(const v16h*)(arow + k0);
        // B frags: lane = column (n0+16t+l15); lanes 0-15 K k0..k0+15, lanes 16-31 +16
        v16h b0 = *(const v16h*)(brow + k0);
        v16h b1 = *(const v16h*)(brow + (size_t)16 * K + k0);
        v16h b2 = *(const v16h*)(brow + (size_t)32 * K + k0);
        v16h b3 = *(const v16h*)(brow + (size_t)48 * K + k0);
        acc0 = __builtin_amdgcn_wmma_f32_16x16x32_f16(false, af, false, b0, (short)0, acc0, false, false);
        acc1 = __builtin_amdgcn_wmma_f32_16x16x32_f16(false, af, false, b1, (short)0, acc1, false, false);
        acc2 = __builtin_amdgcn_wmma_f32_16x16x32_f16(false, af, false, b2, (short)0, acc2, false, false);
        acc3 = __builtin_amdgcn_wmma_f32_16x16x32_f16(false, af, false, b3, (short)0, acc3, false, false);
    }
    int row = m0 + hi * 8;
#pragma unroll
    for (int r = 0; r < 8; ++r) {
        float* crow = C + (size_t)(row + r) * Nc + n0 + l15;
        crow[0]  = acc0[r];
        crow[16] = acc1[r];
        crow[32] = acc2[r];
        crow[48] = acc3[r];
    }
}

// ---------- GAT per-head attention scores ----------
__global__ void gat_scores(const float* __restrict__ h, const float* __restrict__ a_src,
                           const float* __restrict__ a_dst, float* __restrict__ ss,
                           float* __restrict__ sd) {
    int i = blockIdx.x * blockDim.x + threadIdx.x;      // n*4 + head
    if (i >= NN * 4) return;
    int n = i >> 2, hh = i & 3;
    const float* hp = h + (size_t)n * 256 + hh * 64;
    const float* as = a_src + hh * 64;
    const float* ad = a_dst + hh * 64;
    float s1 = 0.f, s2 = 0.f;
#pragma unroll 4
    for (int c = 0; c < 64; ++c) { float v = hp[c]; s1 += v * as[c]; s2 += v * ad[c]; }
    ss[i] = s1; sd[i] = s2;
}

// ---------- segment softmax over destinations ----------
__global__ void attn_max(const int* __restrict__ ei, const float* __restrict__ ss,
                         const float* __restrict__ sd, unsigned* __restrict__ mu) {
    int i = blockIdx.x * blockDim.x + threadIdx.x;      // edge*4 + head
    if (i >= TT * 4) return;
    int e = i >> 2, h = i & 3;
    int src = (e < EE) ? ei[e] : (e - EE);
    int dst = (e < EE) ? ei[EE + e] : (e - EE);
    float ev = leaky(ss[src * 4 + h] + sd[dst * 4 + h]);
    atomicMax(&mu[dst * 4 + h], fkey(ev));
}
__global__ void attn_den(const int* __restrict__ ei, const float* __restrict__ ss,
                         const float* __restrict__ sd, const unsigned* __restrict__ mu,
                         float* __restrict__ den) {
    int i = blockIdx.x * blockDim.x + threadIdx.x;
    if (i >= TT * 4) return;
    int e = i >> 2, h = i & 3;
    int src = (e < EE) ? ei[e] : (e - EE);
    int dst = (e < EE) ? ei[EE + e] : (e - EE);
    float ev = leaky(ss[src * 4 + h] + sd[dst * 4 + h]);
    atomicAdd(&den[dst * 4 + h], __expf(ev - finv(mu[dst * 4 + h])));
}
__global__ void attn_agg(const int* __restrict__ ei, const float* __restrict__ ss,
                         const float* __restrict__ sd, const unsigned* __restrict__ mu,
                         const float* __restrict__ den, const float* __restrict__ h3,
                         float* __restrict__ agg) {
    long long i = (long long)blockIdx.x * blockDim.x + threadIdx.x; // edge*256 + ch
    if (i >= (long long)TT * 256) return;
    int e = (int)(i >> 8), ch = (int)(i & 255), h = ch >> 6;
    int src = (e < EE) ? ei[e] : (e - EE);
    int dst = (e < EE) ? ei[EE + e] : (e - EE);
    float ev = leaky(ss[src * 4 + h] + sd[dst * 4 + h]);
    float alpha = __expf(ev - finv(mu[dst * 4 + h])) / (den[dst * 4 + h] + 1e-16f);
    atomicAdd(&agg[(size_t)dst * 256 + ch], alpha * h3[(size_t)src * 256 + ch]);
}

// ---------- gate: relu(g1 + bgt1) . Wgt2 + bgt2 ----------
__global__ void gate_kernel(const float* __restrict__ g1, const float* __restrict__ bgt1,
                            const float* __restrict__ Wgt2, const float* __restrict__ bgt2,
                            float* __restrict__ gate) {
    __shared__ float s[256];
    int n = blockIdx.x, c = threadIdx.x;
    float v = fmaxf(g1[(size_t)n * 256 + c] + bgt1[c], 0.f);
    s[c] = v * Wgt2[c]; __syncthreads();
    for (int st = 128; st > 0; st >>= 1) { if (c < st) s[c] += s[c + st]; __syncthreads(); }
    if (c == 0) gate[n] = s[0] + bgt2[0];
}

// ---------- global softmax + pooling ----------
__global__ void reduce_max(const float* __restrict__ gate, unsigned* __restrict__ gmax) {
    __shared__ float s[256];
    int i = blockIdx.x * 256 + threadIdx.x;
    s[threadIdx.x] = (i < NN) ? gate[i] : -3.4e38f;
    __syncthreads();
    for (int st = 128; st > 0; st >>= 1) {
        if (threadIdx.x < st) s[threadIdx.x] = fmaxf(s[threadIdx.x], s[threadIdx.x + st]);
        __syncthreads();
    }
    if (threadIdx.x == 0) atomicMax(gmax, fkey(s[0]));
}
__global__ void reduce_den(const float* __restrict__ gate, const unsigned* __restrict__ gmax,
                           float* __restrict__ gden) {
    __shared__ float s[256];
    int i = blockIdx.x * 256 + threadIdx.x;
    float m = finv(*gmax);
    s[threadIdx.x] = (i < NN) ? __expf(gate[i] - m) : 0.f;
    __syncthreads();
    for (int st = 128; st > 0; st >>= 1) {
        if (threadIdx.x < st) s[threadIdx.x] += s[threadIdx.x + st];
        __syncthreads();
    }
    if (threadIdx.x == 0) atomicAdd(gden, s[0]);
}
__global__ void pooled_kernel(const float* __restrict__ x3, const float* __restrict__ gate,
                              const unsigned* __restrict__ gmax, const float* __restrict__ gden,
                              float* __restrict__ pooled) {
    int c = threadIdx.x;                 // channel 0..255
    int n0 = blockIdx.x * 128;
    float m = finv(*gmax), dsum = *gden;
    float acc = 0.f;
    for (int k = 0; k < 128; ++k) {
        int n = n0 + k;
        if (n < NN) {
            float w = __expf(gate[n] - m) / dsum;
            acc += w * x3[(size_t)n * 256 + c];
        }
    }
    atomicAdd(&pooled[c], acc);
}
__global__ void final_out(const float* __restrict__ pooled, const float* __restrict__ Wm,
                          const float* __restrict__ bm, float* __restrict__ out) {
    int c = threadIdx.x;
    float s = bm[c];
    for (int k = 0; k < 256; ++k) s += pooled[k] * Wm[k * 256 + c];
    out[c] = fmaxf(s, 0.f);
}

extern "C" void kernel_launch(void* const* d_in, const int* in_sizes, int n_in,
                              void* d_out, int out_size, void* d_ws, size_t ws_size,
                              hipStream_t stream) {
    (void)in_sizes; (void)n_in; (void)out_size; (void)ws_size;
    const float* x     = (const float*)d_in[0];
    const int*   ei    = (const int*)  d_in[1];
    const float* W1    = (const float*)d_in[2];
    const float* b1    = (const float*)d_in[3];
    const float* W2    = (const float*)d_in[4];
    const float* b2    = (const float*)d_in[5];
    const float* Wg    = (const float*)d_in[6];
    const float* a_src = (const float*)d_in[7];
    const float* a_dst = (const float*)d_in[8];
    const float* bg    = (const float*)d_in[9];
    const float* ln1g  = (const float*)d_in[10];
    const float* ln1b  = (const float*)d_in[11];
    const float* ln2g  = (const float*)d_in[12];
    const float* ln2b  = (const float*)d_in[13];
    const float* ln3g  = (const float*)d_in[14];
    const float* ln3b  = (const float*)d_in[15];
    const float* Wgt1  = (const float*)d_in[16];
    const float* bgt1  = (const float*)d_in[17];
    const float* Wgt2  = (const float*)d_in[18];
    const float* bgt2  = (const float*)d_in[19];
    const float* Wm    = (const float*)d_in[20];
    const float* bm    = (const float*)d_in[21];

    float* outp  = (float*)d_out;        // [256]
    float* x3out = outp + 256;           // [N,256]

    // workspace carve-out (~131 MB, 256B aligned)
    char* w = (char*)d_ws; size_t off = 0;
    auto take = [&](size_t b) -> char* { char* p = w + off; off = (off + b + 255) & ~(size_t)255; return p; };
    float*     dinv   = (float*)    take((size_t)NN * 4);
    float*     Abuf   = (float*)    take((size_t)NN * 256 * 4);   // t1 / t2 / h3 / gate1
    float*     Bbuf   = (float*)    take((size_t)NN * 256 * 4);   // g1 / g2 / agg
    _Float16*  Xh     = (_Float16*) take((size_t)NN * 256 * 2);   // x1h / x2h / x3h
    float*     ssrc   = (float*)    take((size_t)NN * 16);
    float*     sdst   = (float*)    take((size_t)NN * 16);
    unsigned*  mu     = (unsigned*) take((size_t)NN * 16);
    float*     den    = (float*)    take((size_t)NN * 16);
    float*     gate   = (float*)    take((size_t)NN * 4);
    unsigned*  gmax   = (unsigned*) take(4);
    float*     gden   = (float*)    take(4);
    float*     pooled = (float*)    take(256 * 4);
    _Float16*  W2t    = (_Float16*) take(64 * 128 * 2);    // [128,64]
    _Float16*  Wgt    = (_Float16*) take(128 * 256 * 2);   // [256,128]
    _Float16*  Wg1t   = (_Float16*) take(256 * 256 * 2);   // [256,256]

    const int B = 256;
    auto blk = [](long long n, int b) { return (unsigned)((n + b - 1) / b); };

    // --- degrees -> rsqrt ---
    hipMemsetAsync(dinv, 0, (size_t)NN * 4, stream);
    deg_count<<<blk(TT, B), B, 0, stream>>>(ei, dinv);
    dinv_kernel<<<blk(NN, B), B, 0, stream>>>(dinv);

    // --- GCN1: 3 -> 64 ---
    gcn1_mm<<<blk((long long)NN * 64, B), B, 0, stream>>>(x, W1, Abuf);
    hipMemsetAsync(Bbuf, 0, (size_t)NN * 64 * 4, stream);
    gcn_scatter<<<blk((long long)TT * 64, B), B, 0, stream>>>(Abuf, ei, dinv, Bbuf, 64);
    postln<<<NN, 64, 0, stream>>>(Bbuf, b1, ln1g, ln1b, nullptr, Xh, 64, 1);

    // --- GCN2: 64 -> 128 (WMMA) ---
    cvt_tr_f16<<<blk(64 * 128, B), B, 0, stream>>>(W2, W2t, 64, 128);
    wmma_gemm4<<<dim3(NN / 16, 128 / 64), 32, 0, stream>>>(Xh, W2t, Abuf, 64, 128);
    hipMemsetAsync(Bbuf, 0, (size_t)NN * 128 * 4, stream);
    gcn_scatter<<<blk((long long)TT * 128, B), B, 0, stream>>>(Abuf, ei, dinv, Bbuf, 128);
    postln<<<NN, 128, 0, stream>>>(Bbuf, b2, ln2g, ln2b, nullptr, Xh, 128, 1);

    // --- GAT: 128 -> 4x64 (WMMA transform + segment softmax) ---
    cvt_tr_f16<<<blk(128 * 256, B), B, 0, stream>>>(Wg, Wgt, 128, 256);
    wmma_gemm4<<<dim3(NN / 16, 256 / 64), 32, 0, stream>>>(Xh, Wgt, Abuf, 128, 256);
    gat_scores<<<blk((long long)NN * 4, B), B, 0, stream>>>(Abuf, a_src, a_dst, ssrc, sdst);
    hipMemsetAsync(mu,  0, (size_t)NN * 16, stream);
    hipMemsetAsync(den, 0, (size_t)NN * 16, stream);
    hipMemsetAsync(Bbuf, 0, (size_t)NN * 256 * 4, stream);
    attn_max<<<blk((long long)TT * 4, B), B, 0, stream>>>(ei, ssrc, sdst, mu);
    attn_den<<<blk((long long)TT * 4, B), B, 0, stream>>>(ei, ssrc, sdst, mu, den);
    attn_agg<<<blk((long long)TT * 256, B), B, 0, stream>>>(ei, ssrc, sdst, mu, den, Abuf, Bbuf);
    postln<<<NN, 256, 0, stream>>>(Bbuf, bg, ln3g, ln3b, x3out, Xh, 256, 0);

    // --- gate MLP (WMMA) + global softmax pooling ---
    cvt_tr_f16<<<blk(256 * 256, B), B, 0, stream>>>(Wgt1, Wg1t, 256, 256);
    wmma_gemm4<<<dim3(NN / 16, 256 / 64), 32, 0, stream>>>(Xh, Wg1t, Abuf, 256, 256);
    gate_kernel<<<NN, 256, 0, stream>>>(Abuf, bgt1, Wgt2, bgt2, gate);
    hipMemsetAsync(gmax, 0, 4, stream);
    hipMemsetAsync(gden, 0, 4, stream);
    hipMemsetAsync(pooled, 0, 256 * 4, stream);
    reduce_max<<<blk(NN, B), B, 0, stream>>>(gate, gmax);
    reduce_den<<<blk(NN, B), B, 0, stream>>>(gate, gmax, gden);
    pooled_kernel<<<blk(NN, 128), 256, 0, stream>>>(x3out, gate, gmax, gden, pooled);
    final_out<<<1, 256, 0, stream>>>(pooled, Wm, bm, outp);
}
